// SparseMoE_34574486732890
// MI455X (gfx1250) — matrix operand order repaired and checked by reading, not compiled, not run on hardware
//
#include <hip/hip_runtime.h>
#include <hip/hip_bf16.h>

// Problem constants (match reference)
#define B_   8
#define S_   1024
#define D_   1024
#define E_   8
#define H_   4096
#define BS_  (B_ * S_)      // 8192 tokens
#define MT   32             // tokens per FFN block (2 WMMA m-tiles)
#define HC   128            // H-chunk per fused step
#define HSTR (HC + 8)       // padded LDS stride for h (avoid bank conflicts)
#define TILES (BS_ / MT)    // 256 token tiles per expert (worst case)

typedef __attribute__((ext_vector_type(16))) _Float16 v16h;
typedef __attribute__((ext_vector_type(8)))  _Float16 v8h;
typedef __attribute__((ext_vector_type(8)))  float    v8f;

union V16U { v16h v; v8h h[2]; };

__device__ __forceinline__ v8f zero8() {
  v8f z = {0.f, 0.f, 0.f, 0.f, 0.f, 0.f, 0.f, 0.f};
  return z;
}

// ---------------------------------------------------------------------------
// Zero-fill kernel (expert_outputs region must be exactly 0 outside top-k)
// ---------------------------------------------------------------------------
__global__ void zero4_kernel(float* __restrict__ p, size_t n) {
  size_t i = ((size_t)blockIdx.x * blockDim.x + threadIdx.x) * 4;
  if (i < n) *(float4*)(p + i) = make_float4(0.f, 0.f, 0.f, 0.f);
}

__global__ void zero_counts_kernel(int* __restrict__ c) {
  if (threadIdx.x < E_) c[threadIdx.x] = 0;
}

// ---------------------------------------------------------------------------
// Tiled transpose + fp32->f16 convert:  in[R][C] f32 -> out[C][R] f16  (per e)
// grid: (C/32, R/32, E), block: 256
// ---------------------------------------------------------------------------
__global__ __launch_bounds__(256) void transpose_f32_to_f16(
    const float* __restrict__ in, _Float16* __restrict__ out, int R, int C) {
  __shared__ float tile[32][33];
  const int e = blockIdx.z;
  const float* src = in + (size_t)e * R * C;
  _Float16*    dst = out + (size_t)e * R * C;
  const int c0 = blockIdx.x * 32, r0 = blockIdx.y * 32;
  const int tx = threadIdx.x & 31, ty = threadIdx.x >> 5;
#pragma unroll
  for (int i = 0; i < 32; i += 8)
    tile[ty + i][tx] = src[(size_t)(r0 + ty + i) * C + (c0 + tx)];
  __syncthreads();
#pragma unroll
  for (int i = 0; i < 32; i += 8)
    dst[(size_t)(c0 + ty + i) * R + (r0 + tx)] = (_Float16)tile[tx][ty + i];
}

// ---------------------------------------------------------------------------
// Router: one wave32 per token. logits = x@Wg+bg, noisy = logits +
// noise*softplus(x@Wn+bn); top-2 -> softmax -> gating row; append token to
// per-expert compact list.
// ---------------------------------------------------------------------------
__global__ __launch_bounds__(256) void router_kernel(
    const float* __restrict__ x, const float* __restrict__ noise,
    const float* __restrict__ Wg, const float* __restrict__ bg,
    const float* __restrict__ Wn, const float* __restrict__ bn,
    float* __restrict__ gating,          // [BS, E]
    int* __restrict__ counts, int* __restrict__ lists,
    float* __restrict__ gatesl, int* __restrict__ t2e) {
  const int t = blockIdx.x * 8 + (threadIdx.x >> 5);
  const int lane = threadIdx.x & 31;

  float accL[E_], accN[E_];
#pragma unroll
  for (int e = 0; e < E_; ++e) { accL[e] = 0.f; accN[e] = 0.f; }

  const float* xr = x + (size_t)t * D_;
  for (int d0 = lane * 4; d0 < D_; d0 += 128) {
    float4 xv = *(const float4*)(xr + d0);
    const float* xp = (const float*)&xv;
#pragma unroll
    for (int j = 0; j < 4; ++j) {
      const float xs = xp[j];
      const float* wr = Wg + (size_t)(d0 + j) * E_;
      const float* nr = Wn + (size_t)(d0 + j) * E_;
#pragma unroll
      for (int e = 0; e < E_; ++e) {
        accL[e] += xs * wr[e];
        accN[e] += xs * nr[e];
      }
    }
  }
  // wave32 butterfly reduction
#pragma unroll
  for (int off = 16; off > 0; off >>= 1) {
#pragma unroll
    for (int e = 0; e < E_; ++e) {
      accL[e] += __shfl_xor(accL[e], off);
      accN[e] += __shfl_xor(accN[e], off);
    }
  }

  float noisy[E_];
#pragma unroll
  for (int e = 0; e < E_; ++e) {
    float l  = accL[e] + bg[e];
    float nl = accN[e] + bn[e];
    float sp = log1pf(expf(-fabsf(nl))) + fmaxf(nl, 0.f);  // stable softplus
    noisy[e] = l + noise[(size_t)t * E_ + e] * sp;
  }
  // top-2 (first-index tie break, like lax.top_k)
  int i0 = 0;
#pragma unroll
  for (int e = 1; e < E_; ++e) if (noisy[e] > noisy[i0]) i0 = e;
  int i1 = (i0 == 0) ? 1 : 0;
#pragma unroll
  for (int e = 0; e < E_; ++e)
    if (e != i0 && noisy[e] > noisy[i1]) i1 = e;

  const float m  = fmaxf(noisy[i0], noisy[i1]);
  const float e0 = expf(noisy[i0] - m), e1 = expf(noisy[i1] - m);
  const float s  = e0 + e1;
  const float g0 = e0 / s, g1 = e1 / s;

  if (lane < E_)
    gating[(size_t)t * E_ + lane] = (lane == i0) ? g0 : ((lane == i1) ? g1 : 0.f);

  if (lane == 0) {
    int p0 = atomicAdd(&counts[i0], 1);
    lists[i0 * BS_ + p0] = t;  gatesl[i0 * BS_ + p0] = g0;
    int p1 = atomicAdd(&counts[i1], 1);
    lists[i1 * BS_ + p1] = t;  gatesl[i1 * BS_ + p1] = g1;
    t2e[t * 2 + 0] = i0;       t2e[t * 2 + 1] = i1;
  }
}

// ---------------------------------------------------------------------------
// Fused expert FFN: y = relu(x@W1 + b1) @ W2 + b2, gated, scattered.
// grid = E*TILES blocks, 256 threads (8 waves). Each block: 32 gathered
// tokens x full D=1024 output. Per 128-wide H chunk: GEMM1 (each wave makes a
// 32x16 slice of h via v_wmma_f32_16x16x32_f16), relu -> LDS, GEMM2 (each
// wave owns 128 output columns).
// ---------------------------------------------------------------------------
__global__ __launch_bounds__(256) void moe_ffn_kernel(
    const float* __restrict__ x,
    const _Float16* __restrict__ w1t,   // [E,H,D] (K-contiguous columns of W1)
    const _Float16* __restrict__ w2t,   // [E,D,H] (K-contiguous columns of W2)
    const float* __restrict__ b1, const float* __restrict__ b2,
    const int* __restrict__ counts, const int* __restrict__ lists,
    const float* __restrict__ gatesl,
    float* __restrict__ eo)             // [E,BS,D]
{
  const int e    = blockIdx.x / TILES;
  const int tile = blockIdx.x % TILES;
  const int cnt  = counts[e];
  const int m0   = tile * MT;
  if (m0 >= cnt) return;   // uniform early-out; EXEC stays all-1s for WMMA

  extern __shared__ char smem[];
  _Float16* xs   = (_Float16*)smem;                          // MT*D_ halfs, swizzled
  _Float16* hs   = (_Float16*)(smem + (size_t)MT * D_ * 2);  // MT*HSTR halfs
  int*      toks = (int*)(smem + (size_t)MT * D_ * 2 + (size_t)MT * HSTR * 2);
  float*    gts  = (float*)(toks + MT);

  const int tid  = threadIdx.x;
  const int wave = tid >> 5;
  const int lane = tid & 31;
  const int hlf  = lane >> 4;            // half-wave select
  const int ln   = lane & 15;
  const int kofB = hlf ? 16 : 0;         // B-frag: lanes16-31 hold K 16..31
  const int kofA = hlf ? 8  : 0;         // unused; A-frags also use kofB (lanes16-31 hold K 16..31)

  if (tid < MT) {
    int idx = m0 + tid;
    int t = (idx < cnt) ? lists[e * BS_ + idx] : -1;
    toks[tid] = t;
    gts[tid]  = (idx < cnt) ? gatesl[e * BS_ + idx] : 0.f;
  }
  __syncthreads();

  // Stage gathered x tile -> f16 LDS, 8-half groups XOR-swizzled by row
  for (int g = tid; g < MT * (D_ / 8); g += 256) {
    int r  = g >> 7;          // row 0..31
    int kg = g & 127;         // 8-half group 0..127
    int t  = toks[r];
    v8h hv;
    if (t >= 0) {
      const float* src = x + (size_t)t * D_ + kg * 8;
#pragma unroll
      for (int j = 0; j < 8; ++j) hv[j] = (_Float16)src[j];
    } else {
#pragma unroll
      for (int j = 0; j < 8; ++j) hv[j] = (_Float16)0.f;
    }
    *(v8h*)(xs + r * D_ + ((kg ^ (r & 15)) << 3)) = hv;
  }

  const _Float16* w1e = w1t + (size_t)e * H_ * D_;
  const _Float16* w2e = w2t + (size_t)e * D_ * H_;

  v8f yacc[2][8];
#pragma unroll
  for (int mt = 0; mt < 2; ++mt)
#pragma unroll
    for (int nt = 0; nt < 8; ++nt) yacc[mt][nt] = zero8();

  __syncthreads();

  for (int hc = 0; hc < H_; hc += HC) {
    // ---- GEMM1: this wave computes h columns [hc+wave*16, +16), K = D ----
    const int hcol = hc + wave * 16 + ln;
    const _Float16* bptr = w1e + (size_t)hcol * D_;
    v8f hacc0 = zero8(), hacc1 = zero8();
#pragma unroll 4
    for (int kk = 0; kk < D_; kk += 32) {
      v16h bf = *(const v16h*)(bptr + kk + kofB);
#pragma unroll
      for (int mt = 0; mt < 2; ++mt) {
        const int row = mt * 16 + ln;
        const _Float16* base = xs + row * D_;
        const int g0 = ((kk + kofB) >> 3) ^ (row & 15);
        const int g1 = ((kk + kofB + 8) >> 3) ^ (row & 15);
        V16U af;
        af.h[0] = *(const v8h*)(base + (g0 << 3));
        af.h[1] = *(const v8h*)(base + (g1 << 3));
        v8f& hr = mt ? hacc1 : hacc0;
        hr = __builtin_amdgcn_wmma_f32_16x16x32_f16(
            false, af.v, false, bf, (short)0, hr, false, false);
      }
    }
    __syncthreads();  // previous chunk's GEMM2 done reading hs
    {
      const int   ccol = wave * 16 + ln;
      const float b1v  = b1[e * H_ + hc + ccol];
#pragma unroll
      for (int mt = 0; mt < 2; ++mt) {
        const v8f& hr = mt ? hacc1 : hacc0;
        const int rbase = mt * 16 + (hlf ? 8 : 0);
#pragma unroll
        for (int i = 0; i < 8; ++i) {
          float v = hr[i] + b1v;
          v = v > 0.f ? v : 0.f;                 // relu
          hs[(rbase + i) * HSTR + ccol] = (_Float16)v;
        }
      }
    }
    __syncthreads();

    // ---- GEMM2: yacc[:, wave*128..+128] += h_chunk @ W2[hc..+128, cols] ----
#pragma unroll
    for (int kk = 0; kk < HC; kk += 32) {
      V16U af0, af1;
      {
        const _Float16* a0 = hs + (ln)      * HSTR + kk + kofB;
        const _Float16* a1 = hs + (16 + ln) * HSTR + kk + kofB;
        af0.h[0] = *(const v8h*)(a0);      af0.h[1] = *(const v8h*)(a0 + 8);
        af1.h[0] = *(const v8h*)(a1);      af1.h[1] = *(const v8h*)(a1 + 8);
      }
#pragma unroll
      for (int nt = 0; nt < 8; ++nt) {
        const _Float16* bp =
            w2e + (size_t)(wave * 128 + nt * 16 + ln) * H_ + hc + kk + kofB;
        v16h bf = *(const v16h*)bp;
        yacc[0][nt] = __builtin_amdgcn_wmma_f32_16x16x32_f16(
            false, af0.v, false, bf, (short)0, yacc[0][nt], false, false);
        yacc[1][nt] = __builtin_amdgcn_wmma_f32_16x16x32_f16(
            false, af1.v, false, bf, (short)0, yacc[1][nt], false, false);
      }
    }
  }

  // Epilogue: + b2, gate, scatter to expert_outputs
#pragma unroll
  for (int mt = 0; mt < 2; ++mt) {
    const int rbase = mt * 16 + (hlf ? 8 : 0);
    int trow[8]; float grow[8];
#pragma unroll
    for (int i = 0; i < 8; ++i) { trow[i] = toks[rbase + i]; grow[i] = gts[rbase + i]; }
#pragma unroll
    for (int nt = 0; nt < 8; ++nt) {
      const int col = wave * 128 + nt * 16 + ln;
      const float b2v = b2[e * D_ + col];
#pragma unroll
      for (int i = 0; i < 8; ++i) {
        if (trow[i] >= 0) {
          float v = (yacc[mt][nt][i] + b2v) * grow[i];
          eo[((size_t)e * BS_ + trow[i]) * D_ + col] = v;
        }
      }
    }
  }
}

// ---------------------------------------------------------------------------
// final[t] = eo[e0][t] + eo[e1][t]  (each token has exactly top_k=2 experts)
// ---------------------------------------------------------------------------
__global__ __launch_bounds__(256) void final_sum_kernel(
    const float* __restrict__ eo, const int* __restrict__ t2e,
    float* __restrict__ fin) {
  const int t = blockIdx.x;
  const int i = threadIdx.x;  // 256 threads * float4 = 1024 floats
  const int e0 = t2e[t * 2], e1 = t2e[t * 2 + 1];
  const float4* a = (const float4*)(eo + ((size_t)e0 * BS_ + t) * D_);
  const float4* b = (const float4*)(eo + ((size_t)e1 * BS_ + t) * D_);
  float4 va = a[i], vb = b[i];
  float4 r = make_float4(va.x + vb.x, va.y + vb.y, va.z + vb.z, va.w + vb.w);
  ((float4*)(fin + (size_t)t * D_))[i] = r;
}

// ---------------------------------------------------------------------------
extern "C" void kernel_launch(void* const* d_in, const int* in_sizes, int n_in,
                              void* d_out, int out_size, void* d_ws, size_t ws_size,
                              hipStream_t stream) {
  (void)in_sizes; (void)n_in; (void)out_size; (void)ws_size;
  const float* x     = (const float*)d_in[0];
  const float* noise = (const float*)d_in[1];
  const float* Wg    = (const float*)d_in[2];
  const float* bg    = (const float*)d_in[3];
  const float* Wn    = (const float*)d_in[4];
  const float* bn    = (const float*)d_in[5];
  const float* W1    = (const float*)d_in[6];
  const float* b1    = (const float*)d_in[7];
  const float* W2    = (const float*)d_in[8];
  const float* b2    = (const float*)d_in[9];

  float* fin    = (float*)d_out;                 // [BS, D]
  float* eo     = fin + (size_t)BS_ * D_;        // [E, BS, D]
  float* gating = eo + (size_t)E_ * BS_ * D_;    // [BS, E]

  char* ws = (char*)d_ws;
  _Float16* w1t = (_Float16*)ws;                                  // E*H*D f16
  _Float16* w2t = (_Float16*)(ws + (size_t)E_ * H_ * D_ * 2);     // E*D*H f16
  char* ctrl    = ws + (size_t)E_ * H_ * D_ * 4;
  int*   counts = (int*)ctrl;
  int*   lists  = (int*)(ctrl + 256);
  float* gatesl = (float*)(ctrl + 256 + (size_t)E_ * BS_ * 4);
  int*   t2e    = (int*)(ctrl + 256 + (size_t)E_ * BS_ * 8);

  // 1) zero expert_outputs region + per-expert counters
  {
    size_t n = (size_t)E_ * BS_ * D_;
    zero4_kernel<<<dim3((unsigned)(n / 4 / 256)), 256, 0, stream>>>(eo, n);
    zero_counts_kernel<<<1, 32, 0, stream>>>(counts);
  }
  // 2) weights -> f16, transposed so WMMA B-fragments are K-contiguous
  transpose_f32_to_f16<<<dim3(H_ / 32, D_ / 32, E_), 256, 0, stream>>>(W1, w1t, D_, H_);
  transpose_f32_to_f16<<<dim3(D_ / 32, H_ / 32, E_), 256, 0, stream>>>(W2, w2t, H_, D_);
  // 3) router (writes gating output + compact per-expert token lists)
  router_kernel<<<BS_ / 8, 256, 0, stream>>>(x, noise, Wg, bg, Wn, bn,
                                             gating, counts, lists, gatesl, t2e);
  // 4) fused expert FFN (WMMA)
  {
    size_t smem = (size_t)MT * D_ * 2 + (size_t)MT * HSTR * 2 + MT * 4 + MT * 4;
    moe_ffn_kernel<<<E_ * TILES, 256, smem, stream>>>(x, w1t, w2t, b1, b2,
                                                      counts, lists, gatesl, eo);
  }
  // 5) combine: final = sum of the token's two gated expert rows
  final_sum_kernel<<<BS_, 256, 0, stream>>>(eo, t2e, fin);
}
